// coordx_net_63505386438944
// MI455X (gfx1250) — matrix operand order repaired and checked by reading, not compiled, not run on hardware
//
#include <hip/hip_runtime.h>

typedef _Float16 h8   __attribute__((ext_vector_type(8)));
typedef _Float16 v16h __attribute__((ext_vector_type(16)));
typedef float    v8f  __attribute__((ext_vector_type(8)));

#define HRES 1024
#define WRES 1024
#define HID  256

// ---------------------------------------------------------------------------
// Kernel 1: convert qw1 (f32 [256][256], row=N, col=K) -> f16; qw2 (f32 [3][256])
// -> f16 padded to [16][256] (rows 3..15 zero) so layer-2 is one clean N-subtile.
// ---------------------------------------------------------------------------
__global__ void prep_weights(const float* __restrict__ qw1,
                             const float* __restrict__ qw2,
                             _Float16* __restrict__ qw1h,
                             _Float16* __restrict__ qw2h) {
    int idx = blockIdx.x * 256 + threadIdx.x;
    if (idx < HID * HID) qw1h[idx] = (_Float16)qw1[idx];
    if (idx < 16 * HID) {
        int row = idx >> 8;
        qw2h[idx] = (row < 3) ? (_Float16)qw2[idx] : (_Float16)0.0f;
    }
}

// ---------------------------------------------------------------------------
// Kernel 2: per-axis SIREN branch. Blocks 0..1023 -> fx rows, 1024..2047 -> fy.
// h0 = sin(coord*bw + bb); h1 = sin(h0 @ pw1.T + pb1); h2 = sin(h1 @ pw2.T + pb2)
// ---------------------------------------------------------------------------
__global__ void premerge_kernel(const float* __restrict__ x, const float* __restrict__ y,
                                const float* __restrict__ bw0, const float* __restrict__ bb0,
                                const float* __restrict__ bw1, const float* __restrict__ bb1,
                                const float* __restrict__ pw1, const float* __restrict__ pb1,
                                const float* __restrict__ pw2, const float* __restrict__ pb2,
                                float* __restrict__ fx, float* __restrict__ fy) {
    __shared__ __align__(16) float h[HID];
    const int t   = threadIdx.x;
    const int row = blockIdx.x;
    const bool isY = row >= HRES;
    const int r = row & (HRES - 1);

    const float c = isY ? y[r] : x[r];
    const float* w0 = isY ? bw1 : bw0;
    const float* b0 = isY ? bb1 : bb0;
    float* outp = (isY ? fy : fx) + (size_t)r * HID;

    h[t] = __sinf(fmaf(c, w0[t], b0[t]));
    __syncthreads();

    float acc = pb1[t];
    {
        const float4* wrow = (const float4*)(pw1 + (size_t)t * HID);
        const float4* hv   = (const float4*)h;
        #pragma unroll 4
        for (int k = 0; k < HID / 4; ++k) {
            float4 a = wrow[k], b = hv[k];
            acc = fmaf(a.x, b.x, acc); acc = fmaf(a.y, b.y, acc);
            acc = fmaf(a.z, b.z, acc); acc = fmaf(a.w, b.w, acc);
        }
    }
    float h1 = __sinf(acc);
    __syncthreads();
    h[t] = h1;
    __syncthreads();

    acc = pb2[t];
    {
        const float4* wrow = (const float4*)(pw2 + (size_t)t * HID);
        const float4* hv   = (const float4*)h;
        #pragma unroll 4
        for (int k = 0; k < HID / 4; ++k) {
            float4 a = wrow[k], b = hv[k];
            acc = fmaf(a.x, b.x, acc); acc = fmaf(a.y, b.y, acc);
            acc = fmaf(a.z, b.z, acc); acc = fmaf(a.w, b.w, acc);
        }
    }
    outp[t] = __sinf(acc);
}

// ---------------------------------------------------------------------------
// Kernel 3: fused merge -> Linear(256,256)+sin -> Linear(256,3)+sigmoid.
// Workgroup = 8 i-values x 16 j-values (M=128 pairs), 8 waves, wave w owns ii=w.
// A fragments (merged, f16) built in registers; B streamed from L2-hot f16 weights.
// ---------------------------------------------------------------------------
static __device__ __forceinline__ v16h cat16(h8 lo, h8 hi) {
    return __builtin_shufflevector(lo, hi, 0, 1, 2, 3, 4, 5, 6, 7,
                                           8, 9, 10, 11, 12, 13, 14, 15);
}

__global__ void __launch_bounds__(256)
coordx_main(const float* __restrict__ fx, const float* __restrict__ fy,
            const _Float16* __restrict__ qw1h, const float* __restrict__ qb1,
            const _Float16* __restrict__ qw2h, const float* __restrict__ qb2,
            float* __restrict__ out) {
    extern __shared__ char smem_raw[];
    float*    fx_s = (float*)smem_raw;                 //  8 x 256 f32 =  8KB
    float*    fy_s = fx_s + 8 * HID;                   // 16 x 256 f32 = 16KB
    _Float16* z_s  = (_Float16*)(fy_s + 16 * HID);     // 128 x 256 f16 = 64KB

    const int t    = threadIdx.x;
    const int w    = t >> 5;       // wave id 0..7  (== ii within tile)
    const int lane = t & 31;
    const int i0   = blockIdx.x * 8;
    const int j0   = blockIdx.y * 16;

    // Stage fx/fy tiles (coalesced, uniform trip counts -> EXEC stays full).
    for (int idx = t; idx < 8 * HID; idx += 256)
        fx_s[idx] = fx[(size_t)(i0 + (idx >> 8)) * HID + (idx & 255)];
    for (int idx = t; idx < 16 * HID; idx += 256)
        fy_s[idx] = fy[(size_t)(j0 + (idx >> 8)) * HID + (idx & 255)];
    __syncthreads();

    const int  ncol  = lane & 15;          // N (and layer-2 A row) within subtile
    const bool hihalf = (lane >= 16);
    const int  koff0 = hihalf ? 8 : 0;     // A-layout K offsets per ISA table
    const int  koff1 = hihalf ? 24 : 16;
    const int  rbase = hihalf ? 8 : 0;     // C-layout row base

    // ---- Build merged A fragments in registers: a[k] = fx[i]*fy[j] as f16 ----
    v16h afrag[8];
    {
        const float* fxrow = fx_s + w * HID;       // broadcast within wave
        const float* fyrow = fy_s + ncol * HID;
        #pragma unroll
        for (int kk = 0; kk < 8; ++kk) {
            const int kb = kk * 32;
            v16h a;
            #pragma unroll
            for (int u = 0; u < 8; ++u)
                a[u] = (_Float16)(fxrow[kb + koff0 + u] * fyrow[kb + koff0 + u]);
            #pragma unroll
            for (int u = 0; u < 8; ++u)
                a[8 + u] = (_Float16)(fxrow[kb + koff1 + u] * fyrow[kb + koff1 + u]);
            afrag[kk] = a;
        }
    }

    // ---- Layer 1: z = sin(merged @ qw1.T + qb1), stored f16 in per-wave LDS ----
    for (int N = 0; N < 16; ++N) {
        const int n = N * 16 + ncol;
        const _Float16* bp = qw1h + (size_t)n * HID + (hihalf ? 8 : 0);
        v8f acc = {};
        #pragma unroll
        for (int kk = 0; kk < 8; ++kk) {
            h8 blo = *(const h8*)(bp + kk * 32);
            h8 bhi = *(const h8*)(bp + kk * 32 + 16);
            acc = __builtin_amdgcn_wmma_f32_16x16x32_f16(
                false, afrag[kk], false, cat16(blo, bhi), (short)0, acc, false, false);
        }
        const float bias = qb1[n];
        #pragma unroll
        for (int r = 0; r < 8; ++r)
            z_s[(size_t)(w * 16 + rbase + r) * HID + n] = (_Float16)__sinf(acc[r] + bias);
    }
    __syncthreads();

    // ---- Layer 2: out = sigmoid(z @ qw2.T + qb2), N padded to one 16-subtile ----
    {
        const _Float16* zrow = z_s + (size_t)(w * 16 + ncol) * HID;  // A row for this lane
        const _Float16* bp   = qw2h + (size_t)ncol * HID + (hihalf ? 8 : 0);
        v8f acc = {};
        #pragma unroll
        for (int kk = 0; kk < 8; ++kk) {
            const int kb = kk * 32;
            h8 alo = *(const h8*)(zrow + kb + koff0);
            h8 ahi = *(const h8*)(zrow + kb + koff1);
            h8 blo = *(const h8*)(bp + kb);
            h8 bhi = *(const h8*)(bp + kb + 16);
            acc = __builtin_amdgcn_wmma_f32_16x16x32_f16(
                false, cat16(alo, ahi), false, cat16(blo, bhi), (short)0, acc, false, false);
        }
        if (ncol < 3) {                     // only channels 0..2 are real
            const float bias = qb2[ncol];
            #pragma unroll
            for (int r = 0; r < 8; ++r) {
                const int jj = rbase + r;   // pair p = w*16 + jj  -> (i = i0+w, j = j0+jj)
                const float v = acc[r] + bias;
                out[(size_t)ncol * (WRES * HRES) + (size_t)(j0 + jj) * HRES + (i0 + w)] =
                    1.0f / (1.0f + __expf(-v));
            }
        }
    }
}

// ---------------------------------------------------------------------------
extern "C" void kernel_launch(void* const* d_in, const int* in_sizes, int n_in,
                              void* d_out, int out_size, void* d_ws, size_t ws_size,
                              hipStream_t stream) {
    const float* x   = (const float*)d_in[0];
    const float* y   = (const float*)d_in[1];
    const float* bw0 = (const float*)d_in[2];
    const float* bb0 = (const float*)d_in[3];
    const float* bw1 = (const float*)d_in[4];
    const float* bb1 = (const float*)d_in[5];
    const float* pw1 = (const float*)d_in[6];
    const float* pb1 = (const float*)d_in[7];
    const float* pw2 = (const float*)d_in[8];
    const float* pb2 = (const float*)d_in[9];
    const float* qw1 = (const float*)d_in[10];
    const float* qb1 = (const float*)d_in[11];
    const float* qw2 = (const float*)d_in[12];
    const float* qb2 = (const float*)d_in[13];

    char* ws = (char*)d_ws;
    float*    fx   = (float*)(ws);                               // 1 MB
    float*    fy   = (float*)(ws + (1 << 20));                   // 1 MB
    _Float16* qw1h = (_Float16*)(ws + (2 << 20));                // 128 KB
    _Float16* qw2h = (_Float16*)(ws + (2 << 20) + (1 << 17));    // 8 KB

    prep_weights<<<256, 256, 0, stream>>>(qw1, qw2, qw1h, qw2h);
    premerge_kernel<<<2 * HRES, 256, 0, stream>>>(x, y, bw0, bb0, bw1, bb1,
                                                  pw1, pb1, pw2, pb2, fx, fy);

    dim3 grid(HRES / 8, WRES / 16);
    size_t lds_bytes = (8 * HID + 16 * HID) * sizeof(float) + 128 * HID * sizeof(_Float16);
    coordx_main<<<grid, 256, lds_bytes, stream>>>(fx, fy, qw1h, qb1, qw2h, qb2,
                                                  (float*)d_out);
}
